// Block_62483184222773
// MI455X (gfx1250) — matrix operand order repaired and checked by reading, not compiled, not run on hardware
//
#include <hip/hip_runtime.h>

// ---------------------------------------------------------------------------
// MLA block + dense-MoE for MI455X (gfx1250, wave32, WMMA).
// All GEMM work routed through V_WMMA_F32_16X16X32_BF16 (bf16 in, f32 acc).
// GEMM: 64x128 block tile, 32x32 wave tile (4 wmma / k-step), async B staging.
// Flash attention keeps the 4M-entry score matrix in VGPR/LDS.
// ---------------------------------------------------------------------------

typedef unsigned short u16;
typedef __attribute__((ext_vector_type(16))) __bf16 v16bf;
typedef __attribute__((ext_vector_type(8)))  float  v8f;
typedef int v4i __attribute__((ext_vector_type(4)));

#define Bc   2
#define Sc   2048
#define Dm   1024
#define Hh   16
#define DHd  64
#define DRr  16
#define SPL  48
#define DKVc 512
#define FFc  512
#define NEe  8
#define BSt  (Bc*Sc)        // 4096 tokens
#define EPSf 1.1920929e-07f

#if defined(__has_builtin)
#if __has_builtin(__builtin_amdgcn_global_load_async_to_lds_b128)
#define HAVE_ASYNC_LDS 1
#endif
#endif

#define AS1 __attribute__((address_space(1)))
#define AS3 __attribute__((address_space(3)))

// ---- bf16 helpers ---------------------------------------------------------
__device__ __forceinline__ u16 f2b(float f) {
  union { float f; unsigned u; } x; x.f = f;
  unsigned u = x.u;
  u += 0x7fffu + ((u >> 16) & 1u);   // round-to-nearest-even
  return (u16)(u >> 16);
}
__device__ __forceinline__ float b2f(u16 v) {
  union { unsigned u; float f; } x; x.u = ((unsigned)v) << 16; return x.f;
}
// fragment <-> two 128-bit LDS transactions
union frag16 { uint4 q[2]; v16bf v; };

#ifdef HAVE_ASYNC_LDS
__device__ __forceinline__ void wait_async0() {
#if __has_builtin(__builtin_amdgcn_s_wait_asynccnt)
  __builtin_amdgcn_s_wait_asynccnt(0);
#else
  asm volatile("s_wait_asynccnt 0x0" ::: "memory");
#endif
}
__device__ __forceinline__ void async_copy_b128(const u16* gsrc, u16* lds) {
  __builtin_amdgcn_global_load_async_to_lds_b128(
      (AS1 v4i*)(unsigned long long)gsrc,   // global generic == AS1 numerically
      (AS3 v4i*)lds,                        // real addrspacecast generic->local
      0, 0);
}
#endif

#define WMMA_BF16(A, B, C) \
  __builtin_amdgcn_wmma_f32_16x16x32_bf16(false, (A), false, (B), (short)0, (C), false, false)

// ---------------------------------------------------------------------------
// Generic GEMM: C[M,N] = A[M,K](bf16,row) * W[K,N](bf16,row) + bias[N] (f32)
// 256 threads = 8 waves as 2x4; block tile 64x128; wave tile 32x32; K-step 32.
// 4 wmma per wave per k-step (2 A-frags x 2 B-frags) => 2x FLOP per LDS byte.
// ---------------------------------------------------------------------------
__global__ void __launch_bounds__(256) gemm_bf16_wmma(
    const u16* __restrict__ A, const u16* __restrict__ W,
    const float* __restrict__ bias, float* __restrict__ C,
    int M, int N, int K) {
  __shared__ __align__(16) u16 As[64][40];    // 64x32 tile (+pad, 16B rows)
  __shared__ __align__(16) u16 Ws[32][136];   // 32x128 tile (+pad, 16B rows)
  const int tid = threadIdx.x, lane = tid & 31, w = tid >> 5;
  const int wm = w >> 2, wn = w & 3;           // 2x4 wave grid
  const int m0 = blockIdx.y * 64, n0 = blockIdx.x * 128;
  const int ab = (lane >> 4) << 3;             // A-frag k sub-offset
  const int mra = wm * 32 + (lane & 15);       // first A-frag row
  v8f acc00 = {}, acc01 = {}, acc10 = {}, acc11 = {};
  for (int k0 = 0; k0 < K; k0 += 32) {
    __syncthreads();
    // ---- stage A tile: 64x32 = 256 x b128 (one per thread) ----
    {
      int r = tid >> 2, c8 = (tid & 3) << 3;
      *(uint4*)&As[r][c8] = *(const uint4*)&A[(size_t)(m0 + r) * K + k0 + c8];
    }
    // ---- stage W tile: 32x128 = 512 x b128 (two per thread, async) ----
#pragma unroll
    for (int i = 0; i < 2; ++i) {
      int ch = tid + (i << 8);
      int r = ch >> 4, c8 = (ch & 15) << 3;
      const u16* gsrc = &W[(size_t)(k0 + r) * N + n0 + c8];
#ifdef HAVE_ASYNC_LDS
      async_copy_b128(gsrc, &Ws[r][c8]);
#else
      *(uint4*)&Ws[r][c8] = *(const uint4*)gsrc;
#endif
    }
    if (k0 + 32 < K) {   // gfx1250 global_prefetch for next K-tile
      __builtin_prefetch(&A[(size_t)(m0 + (tid & 63)) * K + k0 + 32], 0, 1);
      __builtin_prefetch(&W[(size_t)(k0 + 32 + (tid & 31)) * N + n0], 0, 1);
    }
#ifdef HAVE_ASYNC_LDS
    wait_async0();
#endif
    __syncthreads();
    // ---- fragments: A 2 x ds_load_b128 each; B gathered per k (table order) ----
    frag16 a0, a1, b0, b1;
    a0.q[0] = *(const uint4*)&As[mra][ab];
    a0.q[1] = *(const uint4*)&As[mra][16 + ab];
    a1.q[0] = *(const uint4*)&As[mra + 16][ab];
    a1.q[1] = *(const uint4*)&As[mra + 16][16 + ab];
#pragma unroll
    for (int s = 0; s < 16; ++s) {
      int kk = (s < 8) ? ab + s : 8 + ab + s;
      b0.v[s] = __builtin_bit_cast(__bf16, Ws[kk][wn * 32 + (lane & 15)]);
      b1.v[s] = __builtin_bit_cast(__bf16, Ws[kk][wn * 32 + 16 + (lane & 15)]);
    }
    acc00 = WMMA_BF16(a0.v, b0.v, acc00);
    acc01 = WMMA_BF16(a0.v, b1.v, acc01);
    acc10 = WMMA_BF16(a1.v, b0.v, acc10);
    acc11 = WMMA_BF16(a1.v, b1.v, acc11);
  }
  // C layout: row = vgpr + 8*(lane>=16), col = lane&15
  const int rbase = m0 + wm * 32 + ((lane >> 4) << 3);
  const int c0 = n0 + wn * 32 + (lane & 15);
  const int c1 = c0 + 16;
  const float bi0 = bias ? bias[c0] : 0.f;
  const float bi1 = bias ? bias[c1] : 0.f;
#pragma unroll
  for (int r = 0; r < 8; ++r) {
    C[(size_t)(rbase + r) * N + c0]      = acc00[r] + bi0;
    C[(size_t)(rbase + r) * N + c1]      = acc01[r] + bi1;
    C[(size_t)(rbase + 16 + r) * N + c0] = acc10[r] + bi0;
    C[(size_t)(rbase + 16 + r) * N + c1] = acc11[r] + bi1;
  }
}

// ---------------------------------------------------------------------------
// Flash attention: grid (S/128, B*H); 8 waves x 16 q-rows; k-tiles of 32.
// Q/K/V bf16 in [B,H,S,64]; K transposed in LDS; K/V B-fragments gathered per element.
// ---------------------------------------------------------------------------
__global__ void __launch_bounds__(256) flash_attn(
    const u16* __restrict__ Q, const u16* __restrict__ K,
    const u16* __restrict__ V, float* __restrict__ Of, u16* __restrict__ Ob) {
  __shared__ __align__(16) u16 Kt[64][40];      // transposed: [d][kpos]
  __shared__ __align__(16) u16 Vs[32][72];      // row-major:  [kpos][d]
  __shared__ __align__(16) u16 Ps[8][16][40];   // per-wave P scratch
  const int tid = threadIdx.x, lane = tid & 31, w = tid >> 5;
  const int bh = blockIdx.y, bb = bh >> 4, h = bh & 15;
  const int q0 = blockIdx.x * 128 + w * 16;
  const u16* Qb = Q + (size_t)bh * Sc * DHd;
  const u16* Kb = K + (size_t)bh * Sc * DHd;
  const u16* Vb = V + (size_t)bh * Sc * DHd;
  const int ab = (lane >> 4) << 3;
  const int m = lane & 15;

  // preload Q fragments (two 32-wide d chunks), 4 x global_load_b128
  frag16 qf0, qf1;
  {
    const uint4* qrow = (const uint4*)&Qb[(size_t)(q0 + m) * DHd];
    qf0.q[0] = qrow[ab >> 3];
    qf0.q[1] = qrow[(16 + ab) >> 3];
    qf1.q[0] = qrow[(32 + ab) >> 3];
    qf1.q[1] = qrow[(48 + ab) >> 3];
  }
  float mrow[8], lrow[8];
  v8f o0 = {}, o1 = {}, o2 = {}, o3 = {};
#pragma unroll
  for (int r = 0; r < 8; ++r) { mrow[r] = -3.0e38f; lrow[r] = 0.f; }

  const int kend = blockIdx.x * 128 + 128;   // causal upper bound for block
  const float sc = 0.125f;                   // 1/sqrt(64)
  const int rowoff = (lane >> 4) << 3;
  const int cl = lane & 15;

  for (int kt = 0; kt < kend; kt += 32) {
    __syncthreads();
    {
      int r = tid >> 3, c8 = (tid & 7) << 3;      // r = kpos, c8 = d base
      union { uint4 q; u16 e[8]; } ku;
      ku.q = *(const uint4*)&Kb[(size_t)(kt + r) * DHd + c8];
#pragma unroll
      for (int j = 0; j < 8; ++j) Kt[c8 + j][r] = ku.e[j];   // transpose
      *(uint4*)&Vs[r][c8] = *(const uint4*)&Vb[(size_t)(kt + r) * DHd + c8];
    }
    __syncthreads();

    // ---- scores: S = Q K^T (n-chunks of 16 kpos x d-chunks of 32)
    v8f s0 = {}, s1 = {};
    {
      frag16 bk;
#pragma unroll
      for (int s = 0; s < 16; ++s) bk.v[s] = __builtin_bit_cast(__bf16, Kt[(s < 8 ? ab + s : 8 + ab + s)][cl]);
      s0 = WMMA_BF16(qf0.v, bk.v, s0);
#pragma unroll
      for (int s = 0; s < 16; ++s) bk.v[s] = __builtin_bit_cast(__bf16, Kt[32 + (s < 8 ? ab + s : 8 + ab + s)][cl]);
      s0 = WMMA_BF16(qf1.v, bk.v, s0);
#pragma unroll
      for (int s = 0; s < 16; ++s) bk.v[s] = __builtin_bit_cast(__bf16, Kt[(s < 8 ? ab + s : 8 + ab + s)][16 + cl]);
      s1 = WMMA_BF16(qf0.v, bk.v, s1);
#pragma unroll
      for (int s = 0; s < 16; ++s) bk.v[s] = __builtin_bit_cast(__bf16, Kt[32 + (s < 8 ? ab + s : 8 + ab + s)][16 + cl]);
      s1 = WMMA_BF16(qf1.v, bk.v, s1);
    }

    // ---- online softmax (row reductions across 16 lanes of a half)
#pragma unroll
    for (int r = 0; r < 8; ++r) {
      int qg = q0 + rowoff + r;
      bool v0 = (kt + cl) <= qg;
      bool v1 = (kt + 16 + cl) <= qg;
      float x0 = v0 ? s0[r] * sc : -3.0e38f;
      float x1 = v1 ? s1[r] * sc : -3.0e38f;
      float mx = fmaxf(x0, x1);
#pragma unroll
      for (int off = 8; off > 0; off >>= 1) mx = fmaxf(mx, __shfl_xor(mx, off, 32));
      float mn = fmaxf(mrow[r], mx);
      float corr = __expf(mrow[r] - mn);
      float p0 = v0 ? __expf(x0 - mn) : 0.f;
      float p1 = v1 ? __expf(x1 - mn) : 0.f;
      float rs = p0 + p1;
#pragma unroll
      for (int off = 8; off > 0; off >>= 1) rs += __shfl_xor(rs, off, 32);
      lrow[r] = lrow[r] * corr + rs;
      mrow[r] = mn;
      o0[r] *= corr; o1[r] *= corr; o2[r] *= corr; o3[r] *= corr;
      Ps[w][rowoff + r][cl]      = f2b(p0);
      Ps[w][rowoff + r][16 + cl] = f2b(p1);
    }
    __syncthreads();

    // ---- P (A-layout) + PV accumulation, P via b128 loads, V gathered per element
    frag16 pf, bv;
    pf.q[0] = *(const uint4*)&Ps[w][m][ab];
    pf.q[1] = *(const uint4*)&Ps[w][m][16 + ab];
#pragma unroll
    for (int s = 0; s < 16; ++s) bv.v[s] = __builtin_bit_cast(__bf16, Vs[(s < 8 ? ab + s : 8 + ab + s)][m]);
    o0 = WMMA_BF16(pf.v, bv.v, o0);
#pragma unroll
    for (int s = 0; s < 16; ++s) bv.v[s] = __builtin_bit_cast(__bf16, Vs[(s < 8 ? ab + s : 8 + ab + s)][16 + m]);
    o1 = WMMA_BF16(pf.v, bv.v, o1);
#pragma unroll
    for (int s = 0; s < 16; ++s) bv.v[s] = __builtin_bit_cast(__bf16, Vs[(s < 8 ? ab + s : 8 + ab + s)][32 + m]);
    o2 = WMMA_BF16(pf.v, bv.v, o2);
#pragma unroll
    for (int s = 0; s < 16; ++s) bv.v[s] = __builtin_bit_cast(__bf16, Vs[(s < 8 ? ab + s : 8 + ab + s)][48 + m]);
    o3 = WMMA_BF16(pf.v, bv.v, o3);
  }

  // ---- epilogue: normalize and scatter to [B,S,H*64]
#pragma unroll
  for (int r = 0; r < 8; ++r) {
    float inv = (lrow[r] > 0.f) ? 1.f / lrow[r] : 0.f;
    int row = q0 + rowoff + r;
    size_t ob = ((size_t)bb * Sc + row) * Dm + h * DHd;
    float v0 = o0[r] * inv, v1 = o1[r] * inv, v2 = o2[r] * inv, v3 = o3[r] * inv;
    Of[ob + cl]      = v0; Ob[ob + cl]      = f2b(v0);
    Of[ob + 16 + cl] = v1; Ob[ob + 16 + cl] = f2b(v1);
    Of[ob + 32 + cl] = v2; Ob[ob + 32 + cl] = f2b(v2);
    Of[ob + 48 + cl] = v3; Ob[ob + 48 + cl] = f2b(v3);
  }
}

// ---------------------------------------------------------------------------
// Elementwise / small kernels (vectorized float4 / ushort4)
// ---------------------------------------------------------------------------
__global__ void cvt_kernel(const float* __restrict__ s, u16* __restrict__ d, size_t n) {
  size_t i = (size_t)blockIdx.x * blockDim.x + threadIdx.x;
  size_t st = (size_t)gridDim.x * blockDim.x;
  size_t n4 = n >> 2;
  for (; i < n4; i += st) {
    float4 f = ((const float4*)s)[i];
    ushort4 o;
    o.x = f2b(f.x); o.y = f2b(f.y); o.z = f2b(f.z); o.w = f2b(f.w);
    ((ushort4*)d)[i] = o;
  }
}

__global__ void rmsnorm_k(const float* __restrict__ x, const float* __restrict__ w,
                          u16* __restrict__ ob) {
  const int tok = blockIdx.x; const size_t base = (size_t)tok * Dm;
  const int tid = threadIdx.x;                    // 256 threads, 1 float4 each
  float4 vv = ((const float4*)(x + base))[tid];
  float ss = vv.x * vv.x + vv.y * vv.y + vv.z * vv.z + vv.w * vv.w;
#pragma unroll
  for (int off = 16; off > 0; off >>= 1) ss += __shfl_xor(ss, off, 32);
  __shared__ float red[8]; __shared__ float rinv;
  if ((tid & 31) == 0) red[tid >> 5] = ss;
  __syncthreads();
  if (tid == 0) {
    float t = 0.f;
    for (int i = 0; i < 8; ++i) t += red[i];
    rinv = rsqrtf(t / (float)Dm + EPSf);
  }
  __syncthreads();
  const float r = rinv;
  float4 wv = ((const float4*)w)[tid];
  ushort4 o;
  o.x = f2b(vv.x * r * wv.x); o.y = f2b(vv.y * r * wv.y);
  o.z = f2b(vv.z * r * wv.z); o.w = f2b(vv.w * r * wv.w);
  ((ushort4*)(ob + base))[tid] = o;
}

__global__ void rotary_kernel(const float* __restrict__ src, float* __restrict__ dst) {
  int idx = blockIdx.x * blockDim.x + threadIdx.x;
  if (idx >= BSt * Hh) return;
  int tok = idx >> 4, hh = idx & 15;
  int s = tok & (Sc - 1);
  float t = (float)s / 40.0f;
  size_t base = (size_t)tok * (Hh * DRr) + hh * DRr;
  float xv[16], yv[16];
#pragma unroll
  for (int j = 0; j < 16; ++j) xv[j] = src[base + j];
#pragma unroll
  for (int j = 0; j < 8; ++j) {
    int fi = j & 3;                                   // inv_freq index
    float invf = __powf(10000.0f, -0.25f * (float)fi);
    float ang = t * invf;
    float rot = (j < 4) ? -xv[j + 4] : xv[j - 4];
    yv[j] = xv[j] * __cosf(ang) + rot * __sinf(ang);
  }
#pragma unroll
  for (int j = 8; j < 16; ++j) yv[j] = xv[j];
#pragma unroll
  for (int j = 0; j < 16; ++j) dst[base + j] = yv[j];
}

__global__ void pack_qkv_kernel(const float* __restrict__ qb_, const float* __restrict__ qr_,
                                const float* __restrict__ kb_, const float* __restrict__ kr_,
                                const float* __restrict__ vf_,
                                u16* __restrict__ Q, u16* __restrict__ K, u16* __restrict__ V) {
  int idx = blockIdx.x * blockDim.x + threadIdx.x;
  if (idx >= BSt * Hh * DHd) return;
  int d = idx & 63, t = idx >> 6, hh = t & 15, tok = t >> 4;
  int b = tok >> 11, s = tok & (Sc - 1);
  size_t di = (((size_t)(b * Hh + hh)) * Sc + s) * DHd + d;
  float qv = (d < SPL) ? qb_[(size_t)tok * (Hh * SPL) + hh * SPL + d]
                       : qr_[(size_t)tok * (Hh * DRr) + hh * DRr + (d - SPL)];
  float kv = (d < SPL) ? kb_[(size_t)tok * (Hh * SPL) + hh * SPL + d]
                       : kr_[(size_t)tok * (Hh * DRr) + hh * DRr + (d - SPL)];
  Q[di] = f2b(qv);
  K[di] = f2b(kv);
  V[di] = f2b(vf_[(size_t)tok * Dm + hh * DHd + d]);
}

__global__ void add2_kernel(const float* __restrict__ a, const float* __restrict__ b,
                            float* __restrict__ y, size_t n4) {
  size_t i = (size_t)blockIdx.x * blockDim.x + threadIdx.x;
  size_t st = (size_t)gridDim.x * blockDim.x;
  for (; i < n4; i += st) {
    float4 av = ((const float4*)a)[i], bv = ((const float4*)b)[i];
    float4 yv; yv.x = av.x + bv.x; yv.y = av.y + bv.y;
    yv.z = av.z + bv.z; yv.w = av.w + bv.w;
    ((float4*)y)[i] = yv;
  }
}

__global__ void swiglu_kernel(const float* __restrict__ z, u16* __restrict__ ob) {
  int idx = blockIdx.x * blockDim.x + threadIdx.x;      // over BSt*FFc/4
  if (idx >= BSt * FFc / 4) return;
  int tok = idx >> 7, j4 = idx & 127;
  const float4* row = (const float4*)&z[(size_t)tok * (2 * FFc)];
  float4 a = row[j4], b = row[j4 + 128];
  float4 s;
  s.x = a.x / (1.f + __expf(-a.x)) * b.x;
  s.y = a.y / (1.f + __expf(-a.y)) * b.y;
  s.z = a.z / (1.f + __expf(-a.z)) * b.z;
  s.w = a.w / (1.f + __expf(-a.w)) * b.w;
  ushort4 o; o.x = f2b(s.x); o.y = f2b(s.y); o.z = f2b(s.z); o.w = f2b(s.w);
  ((ushort4*)ob)[idx] = o;
}

__global__ void gate_topk_kernel(const u16* __restrict__ h2, const float* __restrict__ gw,
                                 const float* __restrict__ gb, float* __restrict__ comb) {
  const int tid = threadIdx.x;
  const int tok = blockIdx.x * 32 + (tid >> 3);
  const int e = tid & 7;
  float acc = gb[e];
  size_t base = (size_t)tok * Dm;
  for (int d = 0; d < Dm; d += 4) {
    ushort4 hv = *(const ushort4*)&h2[base + d];
    acc += b2f(hv.x) * gw[(d + 0) * NEe + e];
    acc += b2f(hv.y) * gw[(d + 1) * NEe + e];
    acc += b2f(hv.z) * gw[(d + 2) * NEe + e];
    acc += b2f(hv.w) * gw[(d + 3) * NEe + e];
  }
  float mx = acc;
#pragma unroll
  for (int off = 4; off > 0; off >>= 1) mx = fmaxf(mx, __shfl_xor(mx, off, 32));
  float ex = __expf(acc - mx), sm = ex;
#pragma unroll
  for (int off = 4; off > 0; off >>= 1) sm += __shfl_xor(sm, off, 32);
  float pp = ex / sm;
  int rank = 0;
#pragma unroll
  for (int i = 1; i < 8; ++i) {
    float o = __shfl_xor(pp, i, 32);
    int ol = e ^ i;
    if (o > pp || (o == pp && ol < e)) rank++;
  }
  comb[(size_t)tok * NEe + e] = (rank < 2) ? pp : 0.f;
}

__global__ void axpy_comb_kernel(const float* __restrict__ comb, const float* __restrict__ oe,
                                 float* __restrict__ y, int e) {
  int idx = blockIdx.x * blockDim.x + threadIdx.x;      // over BSt*Dm/4
  if (idx >= BSt * Dm / 4) return;
  int tok = idx >> 8;                                   // 256 float4 per row
  float c = comb[(size_t)tok * NEe + e];
  float4 ov = ((const float4*)oe)[idx];
  float4 yv = ((float4*)y)[idx];
  yv.x += c * ov.x; yv.y += c * ov.y; yv.z += c * ov.z; yv.w += c * ov.w;
  ((float4*)y)[idx] = yv;
}

// ---------------------------------------------------------------------------
// Host-side orchestration
// ---------------------------------------------------------------------------
extern "C" void kernel_launch(void* const* d_in, const int* in_sizes, int n_in,
                              void* d_out, int out_size, void* d_ws, size_t ws_size,
                              hipStream_t stream) {
  (void)in_sizes; (void)n_in; (void)out_size; (void)ws_size;
  const float* x    = (const float*)d_in[0];
  const float* n1w  = (const float*)d_in[1];
  const float* n2w  = (const float*)d_in[2];
  const float* wdkv = (const float*)d_in[3];
  const float* bdkv = (const float*)d_in[4];
  const float* wdq  = (const float*)d_in[5];
  const float* bdq  = (const float*)d_in[6];
  const float* wuk  = (const float*)d_in[7];
  const float* buk  = (const float*)d_in[8];
  const float* wuv  = (const float*)d_in[9];
  const float* buv  = (const float*)d_in[10];
  const float* wuq  = (const float*)d_in[11];
  const float* buq  = (const float*)d_in[12];
  const float* wqr  = (const float*)d_in[13];
  const float* bqr  = (const float*)d_in[14];
  const float* wkr  = (const float*)d_in[15];
  const float* bkr  = (const float*)d_in[16];
  const float* wo   = (const float*)d_in[17];
  const float* bo   = (const float*)d_in[18];
  const float* gw   = (const float*)d_in[19];
  const float* gb   = (const float*)d_in[20];
  const float* ws1  = (const float*)d_in[21];
  const float* bs1  = (const float*)d_in[22];
  const float* ws2  = (const float*)d_in[23];
  const float* bs2  = (const float*)d_in[24];
  const float* wr1  = (const float*)d_in[25];
  const float* br1  = (const float*)d_in[26];
  const float* wr2  = (const float*)d_in[27];
  const float* br2  = (const float*)d_in[28];

  float* out_x   = (float*)d_out;                       // (B,S,D)
  float* out_ckv = out_x + (size_t)BSt * Dm;            // (B,S,DKV)
  float* out_kr  = out_ckv + (size_t)BSt * DKVc;        // (B,S,H,DR)

  char* p = (char*)d_ws;
  auto alloc = [&](size_t bytes) -> char* {
    char* r = p; p += (bytes + 255) & ~(size_t)255; return r;
  };
  // bf16 activation buffers
  u16* hb    = (u16*)alloc((size_t)BSt * Dm * 2);
  u16* h2b   = (u16*)alloc((size_t)BSt * Dm * 2);
  u16* ckvb  = (u16*)alloc((size_t)BSt * DKVc * 2);
  u16* cqb   = (u16*)alloc((size_t)BSt * DKVc * 2);
  u16* attnb = (u16*)alloc((size_t)BSt * Dm * 2);
  u16* z1b   = (u16*)alloc((size_t)BSt * FFc * 2);
  u16* qbuf  = (u16*)alloc((size_t)BSt * Dm * 2);
  u16* kbuf  = (u16*)alloc((size_t)BSt * Dm * 2);
  u16* vbuf  = (u16*)alloc((size_t)BSt * Dm * 2);
  // bf16 weights
  u16* wdkvb = (u16*)alloc((size_t)Dm * DKVc * 2);
  u16* wdqb  = (u16*)alloc((size_t)Dm * DKVc * 2);
  u16* wukb  = (u16*)alloc((size_t)DKVc * Hh * SPL * 2);
  u16* wuvb  = (u16*)alloc((size_t)DKVc * Dm * 2);
  u16* wuqb  = (u16*)alloc((size_t)DKVc * Hh * SPL * 2);
  u16* wqrb  = (u16*)alloc((size_t)DKVc * Hh * DRr * 2);
  u16* wkrb  = (u16*)alloc((size_t)Dm * Hh * DRr * 2);
  u16* wob   = (u16*)alloc((size_t)Dm * Dm * 2);
  u16* ws1b  = (u16*)alloc((size_t)Dm * 2 * FFc * 2);
  u16* ws2b  = (u16*)alloc((size_t)FFc * Dm * 2);
  u16* wr1b  = (u16*)alloc((size_t)NEe * Dm * 2 * FFc * 2);
  u16* wr2b  = (u16*)alloc((size_t)NEe * FFc * Dm * 2);
  // fp32 buffers
  float* cqf    = (float*)alloc((size_t)BSt * DKVc * 4);
  float* kbasef = (float*)alloc((size_t)BSt * Hh * SPL * 4);
  float* vf     = (float*)alloc((size_t)BSt * Dm * 4);
  float* qbasef = (float*)alloc((size_t)BSt * Hh * SPL * 4);
  float* qrotf  = (float*)alloc((size_t)BSt * Hh * DRr * 4);
  float* krotpf = (float*)alloc((size_t)BSt * Hh * DRr * 4);
  float* attnf  = (float*)alloc((size_t)BSt * Dm * 4);
  float* aof    = (float*)alloc((size_t)BSt * Dm * 4);
  float* x1f    = (float*)alloc((size_t)BSt * Dm * 4);
  float* zsf    = (float*)alloc((size_t)BSt * 2 * FFc * 4);
  float* sof    = (float*)alloc((size_t)BSt * Dm * 4);
  float* oef    = (float*)alloc((size_t)BSt * Dm * 4);
  float* combf  = (float*)alloc((size_t)BSt * NEe * 4);

  auto gemm = [&](const u16* A, const u16* W, const float* bias, float* C,
                  int M, int N, int K) {
    dim3 g(N / 128, M / 64);
    gemm_bf16_wmma<<<g, 256, 0, stream>>>(A, W, bias, C, M, N, K);
  };
  auto cvt = [&](const float* s, u16* d, size_t n) {
    cvt_kernel<<<2048, 256, 0, stream>>>(s, d, n);
  };

  // ---- weight conversion to bf16 (deterministic, every launch) ----
  cvt(wdkv, wdkvb, (size_t)Dm * DKVc);
  cvt(wdq,  wdqb,  (size_t)Dm * DKVc);
  cvt(wuk,  wukb,  (size_t)DKVc * Hh * SPL);
  cvt(wuv,  wuvb,  (size_t)DKVc * Dm);
  cvt(wuq,  wuqb,  (size_t)DKVc * Hh * SPL);
  cvt(wqr,  wqrb,  (size_t)DKVc * Hh * DRr);
  cvt(wkr,  wkrb,  (size_t)Dm * Hh * DRr);
  cvt(wo,   wob,   (size_t)Dm * Dm);
  cvt(ws1,  ws1b,  (size_t)Dm * 2 * FFc);
  cvt(ws2,  ws2b,  (size_t)FFc * Dm);
  cvt(wr1,  wr1b,  (size_t)NEe * Dm * 2 * FFc);
  cvt(wr2,  wr2b,  (size_t)NEe * FFc * Dm);

  // ---- attention path ----
  rmsnorm_k<<<BSt, 256, 0, stream>>>(x, n1w, hb);
  gemm(hb, wdkvb, bdkv, out_ckv, BSt, DKVc, Dm);        // c_kv (output #2)
  cvt(out_ckv, ckvb, (size_t)BSt * DKVc);
  gemm(hb, wdqb, bdq, cqf, BSt, DKVc, Dm);
  cvt(cqf, cqb, (size_t)BSt * DKVc);
  gemm(ckvb, wukb, buk, kbasef, BSt, Hh * SPL, DKVc);
  gemm(ckvb, wuvb, buv, vf, BSt, Dm, DKVc);
  gemm(cqb,  wuqb, buq, qbasef, BSt, Hh * SPL, DKVc);
  gemm(cqb,  wqrb, bqr, qrotf, BSt, Hh * DRr, DKVc);
  gemm(hb,   wkrb, bkr, krotpf, BSt, Hh * DRr, Dm);
  {
    int n = BSt * Hh;
    rotary_kernel<<<(n + 255) / 256, 256, 0, stream>>>(qrotf, qrotf);
    rotary_kernel<<<(n + 255) / 256, 256, 0, stream>>>(krotpf, out_kr); // k_rot (output #3)
  }
  {
    int n = BSt * Hh * DHd;
    pack_qkv_kernel<<<(n + 255) / 256, 256, 0, stream>>>(
        qbasef, qrotf, kbasef, out_kr, vf, qbuf, kbuf, vbuf);
  }
  {
    dim3 g(Sc / 128, Bc * Hh);
    flash_attn<<<g, 256, 0, stream>>>(qbuf, kbuf, vbuf, attnf, attnb);
  }
  gemm(attnb, wob, bo, aof, BSt, Dm, Dm);
  {
    size_t n4 = (size_t)BSt * Dm / 4;
    add2_kernel<<<2048, 256, 0, stream>>>(x, aof, x1f, n4);   // x1 = x + attn_out
  }

  // ---- FFN path ----
  rmsnorm_k<<<BSt, 256, 0, stream>>>(x1f, n2w, h2b);
  gemm(h2b, ws1b, bs1, zsf, BSt, 2 * FFc, Dm);
  {
    int n = BSt * FFc / 4;
    swiglu_kernel<<<(n + 255) / 256, 256, 0, stream>>>(zsf, z1b);
  }
  gemm(z1b, ws2b, bs2, sof, BSt, Dm, FFc);
  gate_topk_kernel<<<BSt / 32, 256, 0, stream>>>(h2b, gw, gb, combf);
  {
    size_t n4 = (size_t)BSt * Dm / 4;
    add2_kernel<<<2048, 256, 0, stream>>>(x1f, sof, out_x, n4); // x + shared_out
  }
  for (int e = 0; e < NEe; ++e) {
    gemm(h2b, wr1b + (size_t)e * Dm * 2 * FFc, br1 + (size_t)e * 2 * FFc,
         zsf, BSt, 2 * FFc, Dm);
    {
      int n = BSt * FFc / 4;
      swiglu_kernel<<<(n + 255) / 256, 256, 0, stream>>>(zsf, z1b);
    }
    gemm(z1b, wr2b + (size_t)e * FFc * Dm, br2 + (size_t)e * Dm,
         oef, BSt, Dm, FFc);
    {
      int n = BSt * Dm / 4;
      axpy_comb_kernel<<<(n + 255) / 256, 256, 0, stream>>>(combf, oef, out_x, e);
    }
  }
}
